// AttentionFusion_34402688041245
// MI455X (gfx1250) — matrix operand order repaired and checked by reading, not compiled, not run on hardware
//
#include <hip/hip_runtime.h>

// ---------------------------------------------------------------------------
// Problem constants
// ---------------------------------------------------------------------------
#define B_    16
#define SQ_   2048
#define SK_   2048
#define DQ_   1024
#define DK_   1024
#define MTOT  (B_ * SQ_)          // 32768 rows total
#define DFUS  2048                // DQ_ + DK_

typedef __attribute__((ext_vector_type(16))) __bf16 v16bf;
typedef __attribute__((ext_vector_type(8)))  __bf16 v8bf;
typedef __attribute__((ext_vector_type(8)))  float  v8f;

typedef unsigned int u32x4 __attribute__((ext_vector_type(4)));
typedef int          i32x4 __attribute__((ext_vector_type(4)));
typedef int          i32x8 __attribute__((ext_vector_type(8)));

// Tensor Data Mover availability (arity differs between toolchains)
#if defined(__has_builtin)
#if __has_builtin(__builtin_amdgcn_tensor_load_to_lds) && \
    __has_builtin(__builtin_amdgcn_s_wait_tensorcnt)
#define USE_TDM 1
#endif
#endif

// ---------------------------------------------------------------------------
// WMMA helpers (CDNA5: D = A(16x32) * B(32x16) + C, bf16 in / f32 acc)
// ---------------------------------------------------------------------------
static __device__ __forceinline__ v8f wmma_bf16(v16bf a, v16bf b, v8f c) {
    return __builtin_amdgcn_wmma_f32_16x16x32_bf16(
        /*neg_a=*/false, a, /*neg_b=*/false, b,
        /*c_mod=*/(short)0, c, /*reuse_a=*/false, /*reuse_b=*/false);
}

// A-fragment: row-major [16 x 32] tile, row stride `stride` (elements).
// Lane L (L&15 = row): K chunks [base, base+8) and [base+16, base+24),
// base = (L>>4)*8.   (ISA 7.12.2 "16-bit A-Matrix 16x32")
static __device__ __forceinline__ v16bf load_a_frag(const __bf16* base,
                                                    int stride, int lane) {
    const __bf16* p = base + (size_t)(lane & 15) * stride + ((lane >> 4) << 3);
    v8bf lo = *(const v8bf*)(p);
    v8bf hi = *(const v8bf*)(p + 16);
    v16bf r;
#pragma unroll
    for (int j = 0; j < 8; ++j) { r[j] = lo[j]; r[j + 8] = hi[j]; }
    return r;
}

// B-fragment for A*W^T where W is row-major [N x K]:
// lane L holds W[L&15, kb .. kb+7] and W[L&15, kb+16 .. kb+23], kb = (L>>4)*8.
static __device__ __forceinline__ v16bf load_b_frag(const __bf16* base,
                                                    int stride, int lane) {
    const __bf16* p = base + (size_t)(lane & 15) * stride + ((lane >> 4) << 3);
    v8bf lo = *(const v8bf*)(p);
    v8bf hi = *(const v8bf*)(p + 16);
    v16bf r;
#pragma unroll
    for (int j = 0; j < 8; ++j) { r[j] = lo[j]; r[j + 8] = hi[j]; }
    return r;
}

// ---------------------------------------------------------------------------
// fp32 -> bf16 conversion kernels
// ---------------------------------------------------------------------------
__global__ void cvt_f32_bf16(const float* __restrict__ in,
                             __bf16* __restrict__ out, size_t n) {
    size_t i = (size_t)blockIdx.x * blockDim.x + threadIdx.x;
    size_t stride = (size_t)gridDim.x * blockDim.x;
    for (; i < n; i += stride) out[i] = (__bf16)in[i];
}

// bert [MTOT x 1024] fp32 -> fused[:, 0:1024] bf16 (row stride 2048)
__global__ void cvt_bert_fused(const float* __restrict__ in,
                               __bf16* __restrict__ fused) {
    const size_t n = (size_t)MTOT * DQ_;
    size_t i = (size_t)blockIdx.x * blockDim.x + threadIdx.x;
    size_t stride = (size_t)gridDim.x * blockDim.x;
    for (; i < n; i += stride) {
        size_t r = i >> 10;
        size_t c = i & 1023;
        fused[r * DFUS + c] = (__bf16)in[i];
    }
}

// knowledge [B, SK, DK] fp32 -> kT [B, DK, SK] bf16 (LDS-tiled transpose)
__global__ __launch_bounds__(256) void transpose_kT(
    const float* __restrict__ in, __bf16* __restrict__ out) {
    __shared__ float t[32][33];
    const int b  = blockIdx.z;
    const int s0 = blockIdx.x * 32;
    const int d0 = blockIdx.y * 32;
    const int c = threadIdx.x & 31;
    const int r = threadIdx.x >> 5;  // 0..7
#pragma unroll
    for (int i = 0; i < 4; ++i) {
        int row = r + i * 8;
        t[row][c] = in[((size_t)b * SK_ + s0 + row) * DK_ + d0 + c];
    }
    __syncthreads();
#pragma unroll
    for (int i = 0; i < 4; ++i) {
        int row = r + i * 8;
        out[((size_t)b * DK_ + d0 + row) * SK_ + s0 + c] = (__bf16)t[c][row];
    }
}

// ---------------------------------------------------------------------------
// Tiled WMMA GEMM:  C[M,N] = A[M,K] * W[N,K]^T + bias[N]
// block tile 128x128, 8 waves (2x4), each wave 64x32 (4x2 wmma frags)
// ---------------------------------------------------------------------------
template <bool OUT_BF16>
__global__ __launch_bounds__(256) void gemm_bias_kernel(
    const __bf16* __restrict__ A, int lda,
    const __bf16* __restrict__ W, int ldw,
    const float* __restrict__ bias,
    void* __restrict__ Cout, int ldc, int K) {
    __shared__ __bf16 sA[128 * 32];
    __shared__ __bf16 sW[128 * 32];

    const int tid  = threadIdx.x;
    const int lane = tid & 31;
    const int wave = tid >> 5;
    const int wm   = wave >> 2;   // 0..1 (64-row strip)
    const int wn   = wave & 3;    // 0..3 (32-col strip)
    const size_t m0 = (size_t)blockIdx.x * 128;
    const size_t n0 = (size_t)blockIdx.y * 128;

    v8f acc[4][2];
#pragma unroll
    for (int tm = 0; tm < 4; ++tm)
#pragma unroll
        for (int tn = 0; tn < 2; ++tn)
#pragma unroll
            for (int v = 0; v < 8; ++v) acc[tm][tn][v] = 0.0f;

    const int ldRow  = tid >> 1;         // 0..127
    const int ldHalf = (tid & 1) << 4;   // 0 / 16

    for (int kk = 0; kk < K; kk += 32) {
        *(v16bf*)&sA[ldRow * 32 + ldHalf] =
            *(const v16bf*)&A[(m0 + ldRow) * (size_t)lda + kk + ldHalf];
        *(v16bf*)&sW[ldRow * 32 + ldHalf] =
            *(const v16bf*)&W[(n0 + ldRow) * (size_t)ldw + kk + ldHalf];
        if (kk + 32 < K) {  // prefetch next K slice (global_prefetch_b8)
            __builtin_prefetch(&A[(m0 + ldRow) * (size_t)lda + kk + 32 + ldHalf], 0, 1);
            __builtin_prefetch(&W[(n0 + ldRow) * (size_t)ldw + kk + 32 + ldHalf], 0, 1);
        }
        __syncthreads();

        v16bf bfr[2];
#pragma unroll
        for (int tn = 0; tn < 2; ++tn)
            bfr[tn] = load_b_frag(&sW[(wn * 32 + tn * 16) * 32], 32, lane);
#pragma unroll
        for (int tm = 0; tm < 4; ++tm) {
            v16bf af = load_a_frag(&sA[(wm * 64 + tm * 16) * 32], 32, lane);
            acc[tm][0] = wmma_bf16(af, bfr[0], acc[tm][0]);
            acc[tm][1] = wmma_bf16(af, bfr[1], acc[tm][1]);
        }
        __syncthreads();
    }

    // epilogue: C layout — lane holds col n=(lane&15), rows v + 8*(lane>>4)
#pragma unroll
    for (int tn = 0; tn < 2; ++tn) {
        const size_t col = n0 + wn * 32 + tn * 16 + (lane & 15);
        const float bv = bias[col];
#pragma unroll
        for (int tm = 0; tm < 4; ++tm) {
#pragma unroll
            for (int v = 0; v < 8; ++v) {
                size_t row = m0 + wm * 64 + tm * 16 + v + 8 * (lane >> 4);
                float val = acc[tm][tn][v] + bv;
                if (OUT_BF16)
                    ((__bf16*)Cout)[row * (size_t)ldc + col] = (__bf16)val;
                else
                    ((float*)Cout)[row * (size_t)ldc + col] = val;
            }
        }
    }
}

// ---------------------------------------------------------------------------
// Flash attention: per block = 32 queries of one batch, 8 waves.
// Q tile (32x1024 bf16 = 64KB) staged to LDS via the Tensor Data Mover.
// Key loop in chunks of 64:
//   phase A: each wave one 16x16 S tile via WMMA (K rows contiguous)
//   phase B: online softmax (running max/sum), P re-emitted bf16 in LDS
//   phase C: each wave owns a 128-wide slice of the output; P*V uses the
//            pre-transposed kT so B-fragments are two contiguous 16B loads per lane.
// Output written bf16 into fused[:, 1024:2048].
// ---------------------------------------------------------------------------
__global__ __launch_bounds__(256) void flash_attn_kernel(
    const __bf16* __restrict__ Q, const __bf16* __restrict__ KV,
    const __bf16* __restrict__ KT, __bf16* __restrict__ fused) {
    extern __shared__ char smem[];
    __bf16* sQ = (__bf16*)smem;                               // 64 KB @ offset 0
    float*  sS = (float*)(smem + 65536);                      //  8 KB
    __bf16* sP = (__bf16*)(smem + 65536 + 8192);              //  4 KB
    float*  sM = (float*)(smem + 65536 + 8192 + 4096);        //  stats
    float*  sL = sM + 32;
    float*  sC = sM + 64;

    const int tid  = threadIdx.x;
    const int lane = tid & 31;
    const int wave = tid >> 5;
    const int b    = blockIdx.y;
    const int q0   = blockIdx.x * 32;

    const __bf16* Qb  = Q  + ((size_t)b * SQ_ + q0) * DK_;
    const __bf16* Kb  = KV + (size_t)b * SK_ * DK_;
    const __bf16* kTb = KT + (size_t)b * DK_ * SK_;

#ifdef USE_TDM
    // ---- stage Q tile with the Tensor Data Mover (one DMA from wave 0) ----
    if (wave == 0) {
        unsigned long long ga = (unsigned long long)Qb;
        u32x4 g0;
        g0[0] = 1u;                                   // count=1, user desc
        g0[1] = 0u;                                   // lds_addr = sQ offset 0
        g0[2] = (unsigned)ga;                         // global_addr[31:0]
        g0[3] = (unsigned)((ga >> 32) & 0x01FFFFFFu)  // global_addr[56:32]
                | (2u << 30);                         // type = 2 (image)
        i32x8 g1;
        g1[0] = (int)(1u << 16);        // data_size = 1 (2 bytes)
        g1[1] = (int)(1024u << 16);     // tensor_dim0 = 1024 (bits 63:48)
        g1[2] = (int)(2048u << 16);     // tensor_dim1 = 2048 (bits 111:80 lo)
        g1[3] = (int)(1024u << 16);     // tile_dim0 = 1024 (bits 127:112)
        g1[4] = 32;                     // tile_dim1 = 32
        g1[5] = 1024;                   // tensor_dim0_stride = 1024
        g1[6] = 0;
        g1[7] = 0;
        i32x4 z4 = {0, 0, 0, 0};
#if __clang_major__ >= 23
        i32x8 z8 = {0, 0, 0, 0, 0, 0, 0, 0};
        __builtin_amdgcn_tensor_load_to_lds(g0, g1, z4, z4, z8, 0);
#else
        __builtin_amdgcn_tensor_load_to_lds(g0, g1, z4, z4, 0);
#endif
        __builtin_amdgcn_s_wait_tensorcnt((short)0);
    }
#else
    {   // fallback: cooperative vector copy (32x1024 bf16 = 4096 uint4)
        const uint4* src = (const uint4*)Qb;
        uint4* dst = (uint4*)sQ;
        for (int i = tid; i < 4096; i += 256) dst[i] = src[i];
    }
#endif
    if (tid < 32) { sM[tid] = -1e30f; sL[tid] = 0.0f; }

    v8f acc[2][8];
#pragma unroll
    for (int mt = 0; mt < 2; ++mt)
#pragma unroll
        for (int nt = 0; nt < 8; ++nt)
#pragma unroll
            for (int v = 0; v < 8; ++v) acc[mt][nt][v] = 0.0f;

    __syncthreads();

    const int stm = wave >> 2;  // S tile row (0..1)
    const int stn = wave & 3;   // S tile col (0..3)

    for (int kc = 0; kc < SK_; kc += 64) {
        // ---- phase A: S = Q * K^T (this wave's 16x16 tile) ----
        v8f s;
#pragma unroll
        for (int v = 0; v < 8; ++v) s[v] = 0.0f;
        for (int kk = 0; kk < DK_; kk += 32) {
            v16bf af = load_a_frag(sQ + (size_t)(stm * 16) * DK_ + kk, DK_, lane);
            v16bf bf = load_b_frag(Kb + (size_t)(kc + stn * 16) * DK_ + kk, DK_, lane);
            s = wmma_bf16(af, bf, s);
        }
        {
            const int col = stn * 16 + (lane & 15);
#pragma unroll
            for (int v = 0; v < 8; ++v) {
                int row = stm * 16 + v + 8 * (lane >> 4);
                sS[row * 64 + col] = s[v] * 0.03125f;  // 1/sqrt(1024)
            }
        }
        __syncthreads();

        // ---- phase B: online softmax (32 threads, one row each) ----
        if (tid < 32) {
            const float* srow = sS + tid * 64;
            float mold = sM[tid];
            float mx = mold;
#pragma unroll 8
            for (int c = 0; c < 64; ++c) mx = fmaxf(mx, srow[c]);
            float corr = __expf(mold - mx);
            float l = sL[tid] * corr;
            __bf16* prow = sP + tid * 64;
#pragma unroll 8
            for (int c = 0; c < 64; ++c) {
                float e = __expf(srow[c] - mx);
                prow[c] = (__bf16)e;
                l += e;
            }
            sM[tid] = mx; sL[tid] = l; sC[tid] = corr;
        }
        __syncthreads();

        // ---- phase C: rescale O, accumulate P*V for this wave's slice ----
#pragma unroll
        for (int mt = 0; mt < 2; ++mt) {
            float cf[8];
#pragma unroll
            for (int v = 0; v < 8; ++v)
                cf[v] = sC[mt * 16 + v + 8 * (lane >> 4)];
#pragma unroll
            for (int nt = 0; nt < 8; ++nt)
#pragma unroll
                for (int v = 0; v < 8; ++v) acc[mt][nt][v] *= cf[v];
        }
#pragma unroll
        for (int kk2 = 0; kk2 < 2; ++kk2) {
            v16bf pa[2];
#pragma unroll
            for (int mt = 0; mt < 2; ++mt)
                pa[mt] = load_a_frag(sP + mt * 16 * 64 + kk2 * 32, 64, lane);
#pragma unroll
            for (int nt = 0; nt < 8; ++nt) {
                // V^T rows are contiguous in kT: two 16B loads per lane.
                v16bf bf = load_b_frag(
                    kTb + (size_t)(wave * 128 + nt * 16) * SK_ + kc + kk2 * 32,
                    SK_, lane);
                acc[0][nt] = wmma_bf16(pa[0], bf, acc[0][nt]);
                acc[1][nt] = wmma_bf16(pa[1], bf, acc[1][nt]);
            }
        }
        __syncthreads();
    }

    // ---- epilogue: O /= l, write bf16 into fused[:, 1024:2048] ----
#pragma unroll
    for (int mt = 0; mt < 2; ++mt) {
        float inv[8];
#pragma unroll
        for (int v = 0; v < 8; ++v)
            inv[v] = 1.0f / sL[mt * 16 + v + 8 * (lane >> 4)];
#pragma unroll
        for (int nt = 0; nt < 8; ++nt) {
            const int col = wave * 128 + nt * 16 + (lane & 15);
#pragma unroll
            for (int v = 0; v < 8; ++v) {
                size_t qrow = (size_t)q0 + mt * 16 + v + 8 * (lane >> 4);
                fused[((size_t)b * SQ_ + qrow) * DFUS + DQ_ + col] =
                    (__bf16)(acc[mt][nt][v] * inv[v]);
            }
        }
    }
}

// ---------------------------------------------------------------------------
// Launch
// ---------------------------------------------------------------------------
extern "C" void kernel_launch(void* const* d_in, const int* in_sizes, int n_in,
                              void* d_out, int out_size, void* d_ws,
                              size_t ws_size, hipStream_t stream) {
    const float* bert = (const float*)d_in[0];
    const float* know = (const float*)d_in[1];
    const float* w1w  = (const float*)d_in[2];
    const float* w1b  = (const float*)d_in[3];
    const float* w2w  = (const float*)d_in[4];
    const float* w2b  = (const float*)d_in[5];
    float* out = (float*)d_out;

    // ws carve (bf16): fused 128MB | K 64MB | kT 64MB | Q 64MB | w1 2MB | w2 4MB
    char* ws = (char*)d_ws;
    __bf16* fused = (__bf16*)ws; ws += (size_t)MTOT * DFUS * 2;
    __bf16* kbf   = (__bf16*)ws; ws += (size_t)B_ * SK_ * DK_ * 2;
    __bf16* kT    = (__bf16*)ws; ws += (size_t)B_ * DK_ * SK_ * 2;
    __bf16* qbf   = (__bf16*)ws; ws += (size_t)MTOT * DK_ * 2;
    __bf16* w1bf  = (__bf16*)ws; ws += (size_t)DK_ * DQ_ * 2;
    __bf16* w2bf  = (__bf16*)ws; ws += (size_t)DQ_ * DFUS * 2;

    cvt_bert_fused<<<2048, 256, 0, stream>>>(bert, fused);
    cvt_f32_bf16<<<2048, 256, 0, stream>>>(know, kbf, (size_t)B_ * SK_ * DK_);
    dim3 gt(SK_ / 32, DK_ / 32, B_);
    transpose_kT<<<gt, 256, 0, stream>>>(know, kT);
    cvt_f32_bf16<<<64, 256, 0, stream>>>(w1w, w1bf, (size_t)DK_ * DQ_);
    cvt_f32_bf16<<<128, 256, 0, stream>>>(w2w, w2bf, (size_t)DQ_ * DFUS);

    // GEMM1: Q = bert * w1^T + b1   (bf16 out)
    dim3 g1(MTOT / 128, DK_ / 128);
    gemm_bias_kernel<true><<<g1, 256, 0, stream>>>(fused, DFUS, w1bf, DQ_,
                                                   w1b, qbf, DK_, DQ_);

    // Flash attention -> fused[:, 1024:2048]
    dim3 gf(SQ_ / 32, B_);
    size_t shmem = 65536 + 8192 + 4096 + 3 * 32 * sizeof(float);
    flash_attn_kernel<<<gf, 256, shmem, stream>>>(qbf, kbf, kT, fused);

    // GEMM2: out = fused * w2^T + b2   (fp32 out)
    dim3 g2(MTOT / 128, DQ_ / 128);
    gemm_bias_kernel<false><<<g2, 256, 0, stream>>>(fused, DFUS, w2bf, DFUS,
                                                    w2b, out, DQ_, DFUS);
}
